// Generator_40218073759784
// MI455X (gfx1250) — matrix operand order repaired and checked by reading, not compiled, not run on hardware
//
#include <hip/hip_runtime.h>
#include <hip/hip_bf16.h>

// ---------------------------------------------------------------------------
// Seq2seq GRU for MI455X / gfx1250.  All matmuls on v_wmma_f32_16x16x32_f16.
// One persistent kernel runs the 2047 sequential steps.  Each workgroup owns a
// (16-wide hidden-column slice, batch-half); gate pre-activations live in LDS
// so GEMM + GRU update fuse into one phase (workgroup-local sync only).
// Weight B-fragments are staged into LDS with global_load_async_to_lds.
// ---------------------------------------------------------------------------

typedef _Float16 half_t;
typedef __attribute__((ext_vector_type(16))) _Float16 v16h;
typedef __attribute__((ext_vector_type(8)))  float    v8f;
typedef int b128i __attribute__((vector_size(16)));   // matches builtin sig

#define NB  64          // persistent grid: 64 WGs x 256 threads = 512 waves
#define BTH 256

#define BSZ 128
#define TSEQ 1024
#define VDIM 390
#define EDIM 256
#define HDIM 512
#define G3H  1536
#define START_TOK 388

#if __has_builtin(__builtin_amdgcn_global_load_async_to_lds_b128)
#define HAS_ASYNC_LDS 1
#endif

// ---------------- fragment index helpers ----------------------------------
// A matrix (16x32 f16, ISA 7.12.2): element i of lane's v16h:
//   K = i + 8*(lane>>4) + (i>=8 ? 8 : 0); fragments stored linearly as
//   ((mt*KT + kt)*32 + lane)*16 + i.
__device__ __host__ inline int fragA_index(int b, int j, int KT) {
  int mt = b >> 4, lane_lo = b & 15;
  int kt = j >> 5, kk = j & 31;
  int khalf, i;
  if (kk < 8)       { khalf = 0; i = kk;      }
  else if (kk < 16) { khalf = 1; i = kk - 8;  }
  else if (kk < 24) { khalf = 0; i = kk - 8;  }
  else              { khalf = 1; i = kk - 16; }
  return (((mt * KT + kt) * 32) + lane_lo + 16 * khalf) * 16 + i;
}

// ---------------- WMMA K-loop ----------------------------------------------
__device__ __forceinline__ v8f wmma_kloop(const half_t* ap, const half_t* bp,
                                          int KT) {
  v8f acc = {};
  for (int kt = 0; kt < KT; ++kt) {
    v16h av = *(const v16h*)ap;
    v16h bv = *(const v16h*)bp;
    acc = __builtin_amdgcn_wmma_f32_16x16x32_f16(
        false, av, false, bv, (short)0, acc, false, false);
    ap += 512; bp += 512;
  }
  return acc;
}

// ---------------- async global -> LDS staging ------------------------------
__device__ __forceinline__ void copy16_to_lds(const char* src, char* ldst) {
#if defined(HAS_ASYNC_LDS)
  __builtin_amdgcn_global_load_async_to_lds_b128(
      (__attribute__((address_space(1))) b128i*)(char*)src,
      (__attribute__((address_space(3))) b128i*)ldst, 0, 0);
#else
  *(int4*)ldst = *(const int4*)src;       // global_load_b128 + ds_store_b128
#endif
}

__device__ __forceinline__ void async_wait_all() {
#if defined(HAS_ASYNC_LDS)
#if __has_builtin(__builtin_amdgcn_s_wait_asynccnt)
  __builtin_amdgcn_s_wait_asynccnt(0);
#else
  asm volatile("s_wait_asynccnt 0" ::: "memory");
#endif
#endif
}

// Stage the 3 gate-column blocks (nt = s, s+32, s+64) of a packed weight
// into LDS.  Each gate block is KT*1024 contiguous bytes.
__device__ __forceinline__ void stage_B(const half_t* pw, int KT, int s,
                                        char* ldsdst, int tidb) {
  int chunks = 3 * KT * 64;               // 16B chunks
  for (int c = tidb; c < chunks; c += BTH) {
    int g  = c / (KT * 64);
    int ci = c % (KT * 64);
    const char* src = (const char*)pw +
        ((size_t)(s + 32 * g) * KT) * 1024 + (size_t)ci * 16;
    char* dst = ldsdst + (size_t)g * KT * 1024 + (size_t)ci * 16;
    copy16_to_lds(src, dst);
  }
}

// ---------------- device-wide sense barrier --------------------------------
__device__ inline void gbar(int* arrive, int* release, int& phase) {
  __syncthreads();
  if (threadIdx.x == 0) {
    __threadfence();
    int old = __hip_atomic_fetch_add(arrive, 1, __ATOMIC_ACQ_REL,
                                     __HIP_MEMORY_SCOPE_AGENT);
    if (old == NB - 1) {
      __hip_atomic_store(arrive, 0, __ATOMIC_RELAXED, __HIP_MEMORY_SCOPE_AGENT);
      __hip_atomic_store(release, phase, __ATOMIC_RELEASE, __HIP_MEMORY_SCOPE_AGENT);
    } else {
      while (__hip_atomic_load(release, __ATOMIC_ACQUIRE,
                               __HIP_MEMORY_SCOPE_AGENT) < phase) {
        __builtin_amdgcn_s_sleep(2);
      }
    }
    __threadfence();
  }
  __syncthreads();
  ++phase;
}

// ---------------- fused GRU layer phase ------------------------------------
// WG (s, m2) computes gi/gh for j in [16s,16s+16), b in [64*m2, 64*m2+64),
// keeps gates in LDS, then updates h in-place.  AgiBase is pre-offset to the
// m2 batch-half (global) or a WG-local LDS buffer; indexed by mt_local.
__device__ __forceinline__ void gru_layer(
    const half_t* AgiBase, int KTi,
    const half_t* __restrict__ pwBih, const half_t* __restrict__ pwBhh,
    const float* __restrict__ bih, const float* __restrict__ bhh,
    const half_t* __restrict__ hin, half_t* __restrict__ hout,
    float* __restrict__ h32,
    char* lds, int eA_bytes,
    int s, int m2, int wv, int lane, int tidb) {
  const int off_Bgi   = 0;
  const int off_Bgh   = KTi * 3 * 1024;
  const int off_gates = off_Bgh + 48 * 1024 + eA_bytes;

  stage_B(pwBih, KTi, s, lds + off_Bgi, tidb);
  stage_B(pwBhh, 16,  s, lds + off_Bgh, tidb);
  async_wait_all();
  __syncthreads();

  // compute: wave wv -> hsel = wv>>2 (0: gi, 1: gh), mt_local = wv&3
  int hsel = wv >> 2, mt_local = wv & 3;
  const half_t* Af = hsel ? (hin + (size_t)(m2 * 4) * 16 * 512) : AgiBase;
  int KT = hsel ? 16 : KTi;
  const half_t* BL = (const half_t*)(lds + (hsel ? off_Bgh : off_Bgi));
  float* gatesL = (float*)(lds + off_gates) + hsel * 3072;
  const half_t* ap0 = Af + (size_t)mt_local * KT * 512 + lane * 16;
  int jj = lane & 15;
  int rb = (lane >> 4) << 3;
  for (int g = 0; g < 3; ++g) {
    v8f acc = wmma_kloop(ap0, BL + (size_t)g * KT * 512 + lane * 16, KT);
#pragma unroll
    for (int r = 0; r < 8; ++r) {
      int b_local = 16 * mt_local + rb + r;
      gatesL[(b_local * 3 + g) * 16 + jj] = acc[r];
    }
  }
  __syncthreads();

  // fused GRU update for this WG's (b, j) slice
  const float* giL = (const float*)(lds + off_gates);
  const float* ghL = giL + 3072;
  for (int idx = tidb; idx < 1024; idx += BTH) {
    int b_local = idx >> 4, j16 = idx & 15;
    int j = s * 16 + j16, b = m2 * 64 + b_local;
    const float* gp = giL + (b_local * 3) * 16 + j16;
    const float* hp = ghL + (b_local * 3) * 16 + j16;
    float r  = gp[0]  + bih[j]        + hp[0]  + bhh[j];
    float z  = gp[16] + bih[512 + j]  + hp[16] + bhh[512 + j];
    float nn = gp[32] + bih[1024 + j];
    float hn = hp[32] + bhh[1024 + j];
    r = 1.f / (1.f + __expf(-r));
    z = 1.f / (1.f + __expf(-z));
    float n = tanhf(nn + r * hn);
    size_t hidx = (size_t)b * 512 + j;
    float hnew = (1.f - z) * n + z * h32[hidx];
    h32[hidx] = hnew;
    hout[fragA_index(b, j, 16)] = (half_t)hnew;
  }
}

// ---------------- kernel arg bundle ----------------------------------------
struct SeqArgs {
  const half_t* xfrag;
  half_t *h0fA, *h0fB, *h1fA, *h1fB, *hidf;
  float *h0_32, *h1_32, *logits;
  const half_t *pw_eWih0, *pw_eWhh0, *pw_eWih1, *pw_eWhh1;
  const half_t *pw_dWih0, *pw_dWhh0, *pw_dWih1, *pw_dWhh1;
  const half_t *pw_W1, *pw_W2;
  const float *ebih0, *ebhh0, *ebih1, *ebhh1;
  const float *dbih0, *dbhh0, *dbih1, *dbhh1;
  const float *dec_emb, *b1, *b2;
  int *tok, *bar;
  float *outp, *maxout;
};

// ---------------- persistent sequential kernel -----------------------------
__global__ __launch_bounds__(BTH) void seq_kernel(SeqArgs a) {
  __shared__ __align__(16) char lds[128 * 1024];
  const int tidb = threadIdx.x;
  const int wv   = tidb >> 5;
  const int lane = tidb & 31;
  const int wg   = blockIdx.x;          // 0..63
  const int s    = wg >> 1;             // hidden-column slice 0..31
  const int m2   = wg & 1;              // batch half
  const int wave = wg * 8 + wv;
  const int nwaves = NB * 8;
  int phase = 1;
  int* arrive  = a.bar;
  int* release = a.bar + 1;

  half_t* h0cur = a.h0fA; half_t* h0nxt = a.h0fB;
  half_t* h1cur = a.h1fA; half_t* h1nxt = a.h1fB;

  // ===================== encoder =====================
  for (int t = 0; t < TSEQ; ++t) {
    const half_t* xf = a.xfrag + (size_t)t * (BSZ * EDIM) +
                       (size_t)(m2 * 4) * 8 * 512;
    gru_layer(xf, 8, a.pw_eWih0, a.pw_eWhh0, a.ebih0, a.ebhh0,
              h0cur, h0nxt, a.h0_32, lds, 0, s, m2, wv, lane, tidb);
    gbar(arrive, release, phase);
    gru_layer(h0nxt + (size_t)(m2 * 4) * 16 * 512, 16,
              a.pw_eWih1, a.pw_eWhh1, a.ebih1, a.ebhh1,
              h1cur, h1nxt, a.h1_32, lds, 0, s, m2, wv, lane, tidb);
    gbar(arrive, release, phase);
    { half_t* tmp = h0cur; h0cur = h0nxt; h0nxt = tmp; }
    { half_t* tmp = h1cur; h1cur = h1nxt; h1nxt = tmp; }
  }

  // ===================== decoder =====================
  const int off_eA = 8 * 3 * 1024 + 48 * 1024;      // 73728: after Bgi(24K)+Bgh(48K)
  half_t* eL = (half_t*)(lds + off_eA);
  for (int t = 1; t < TSEQ; ++t) {
    // gather this batch-half's token embeddings into LDS A-fragments (KT=8)
    for (int idx = tidb; idx < 64 * EDIM; idx += BTH) {
      int b_local = idx >> 8, j = idx & 255;
      int b = m2 * 64 + b_local;
      eL[fragA_index(b_local, j, 8)] =
          (half_t)a.dec_emb[(size_t)a.tok[b] * EDIM + j];
    }
    gru_layer(eL, 8, a.pw_dWih0, a.pw_dWhh0, a.dbih0, a.dbhh0,
              h0cur, h0nxt, a.h0_32, lds, 32 * 1024, s, m2, wv, lane, tidb);
    gbar(arrive, release, phase);
    gru_layer(h0nxt + (size_t)(m2 * 4) * 16 * 512, 16,
              a.pw_dWih1, a.pw_dWhh1, a.dbih1, a.dbhh1,
              h1cur, h1nxt, a.h1_32, lds, 0, s, m2, wv, lane, tidb);
    gbar(arrive, release, phase);
    { half_t* tmp = h0cur; h0cur = h0nxt; h0nxt = tmp; }
    { half_t* tmp = h1cur; h1cur = h1nxt; h1nxt = tmp; }

    // hid = leaky(h1 @ W1^T + b1) -> A-fragments (fused epilogue)
    if (wv < 4) {
      int mt = m2 * 4 + wv;
      const half_t* ap = h1cur + (size_t)mt * 16 * 512 + lane * 16;
      const half_t* bp = a.pw_W1 + (size_t)s * 16 * 512 + lane * 16;
      v8f acc = wmma_kloop(ap, bp, 16);
      int n = 16 * s + (lane & 15);
      int rb = (lane >> 4) << 3;
      float bb = a.b1[n];
#pragma unroll
      for (int r = 0; r < 8; ++r) {
        float v = acc[r] + bb;
        v = (v >= 0.f) ? v : 0.1f * v;
        a.hidf[fragA_index(16 * mt + rb + r, n, 16)] = (half_t)v;
      }
    }
    gbar(arrive, release, phase);

    // logits = hid @ W2^T + b2  (N padded to 400, zero-packed weights)
    for (int p = wave; p < 200; p += nwaves) {
      int mt = p & 7, nt = p >> 3;
      const half_t* ap = a.hidf + (size_t)mt * 16 * 512 + lane * 16;
      const half_t* bp = a.pw_W2 + (size_t)nt * 16 * 512 + lane * 16;
      v8f acc = wmma_kloop(ap, bp, 16);
      int n = 16 * nt + (lane & 15);
      int rb = (lane >> 4) << 3;
      float bb = a.b2[(n < VDIM) ? n : (VDIM - 1)];
      bb = (n < VDIM) ? bb : 0.f;
#pragma unroll
      for (int r = 0; r < 8; ++r)
        a.logits[(size_t)(16 * mt + rb + r) * 400 + n] = acc[r] + bb;
    }
    gbar(arrive, release, phase);

    // softmax + argmax: one wave32 per batch row
    if (wave < BSZ) {
      int b = wave;
      const float* lr = a.logits + (size_t)b * 400;
      float vmax = -__builtin_inff(); int vidx = 0x7fffffff;
      for (int c = lane; c < VDIM; c += 32) {
        float v = lr[c];
        if (v > vmax || (v == vmax && c < vidx)) { vmax = v; vidx = c; }
      }
      for (int off = 16; off; off >>= 1) {
        float ov = __shfl_xor(vmax, off, 32);
        int   oi = __shfl_xor(vidx, off, 32);
        if (ov > vmax || (ov == vmax && oi < vidx)) { vmax = ov; vidx = oi; }
      }
      float ssum = 0.f;
      for (int c = lane; c < VDIM; c += 32) ssum += __expf(lr[c] - vmax);
      for (int off = 16; off; off >>= 1) ssum += __shfl_xor(ssum, off, 32);
      float inv = 1.f / ssum;
      float* orow = a.outp + ((size_t)b * TSEQ + t) * VDIM;
      for (int c = lane; c < VDIM; c += 32)
        orow[c] = __expf(lr[c] - vmax) * inv;
      if (lane == 0) {
        a.tok[b] = vidx;
        a.maxout[(size_t)b * TSEQ + t] = (float)vidx;
      }
    }
    gbar(arrive, release, phase);
  }
}

// ---------------- weight pack: f32 W[N,K] -> WMMA B fragments (f16) --------
// B (32x16 f16): lane 0-15 col N, K = (i&7) + 8*(lane>>4) + (i>=8 ? 16 : 0).  Zero-pad OOB.
__global__ __launch_bounds__(BTH) void pack_b_kernel(
    const float* __restrict__ W, half_t* __restrict__ dst,
    int Nreal, int Kreal, int NT, int KT) {
  size_t total = (size_t)NT * KT * 512;
  for (size_t idx = (size_t)blockIdx.x * BTH + threadIdx.x; idx < total;
       idx += (size_t)gridDim.x * BTH) {
    int    i    = (int)(idx & 15);
    int    lane = (int)((idx >> 4) & 31);
    size_t tile = idx >> 9;
    int    kt   = (int)(tile % KT);
    int    nt   = (int)(tile / KT);
    int kk = (i & 7) + 8 * (lane >> 4) + (i >= 8 ? 16 : 0);
    int n  = 16 * nt + (lane & 15);
    int K  = 32 * kt + kk;
    float v = (n < Nreal && K < Kreal) ? W[(size_t)n * Kreal + K] : 0.f;
    dst[idx] = (half_t)v;
  }
}

// ---------------- embedding GEMM: x = input @ enc_emb ----------------------
__global__ __launch_bounds__(BTH) void embed_kernel(
    const float* __restrict__ input, const float* __restrict__ emb,
    half_t* __restrict__ xfrag) {
  __shared__ float row[VDIM];
  size_t bt = blockIdx.x;                 // = b*TSEQ + t
  int b = (int)(bt >> 10), t = (int)(bt & 1023);
  const float* in = input + bt * (size_t)VDIM;
  for (int v = threadIdx.x; v < VDIM; v += BTH) row[v] = in[v];
  __syncthreads();
  int e = threadIdx.x;                    // 256 threads == EDIM
  float acc = 0.f;
  for (int v = 0; v < VDIM; ++v)
    acc = fmaf(row[v], emb[(size_t)v * EDIM + e], acc);
  xfrag[(size_t)t * (BSZ * EDIM) + fragA_index(b, e, 8)] = (half_t)acc;
}

// ---------------- init: state, barrier, t=0 outputs ------------------------
__global__ __launch_bounds__(BTH) void init_kernel(
    float* h0_32, float* h1_32,
    half_t* h0fA, half_t* h0fB, half_t* h1fA, half_t* h1fB,
    int* tok, int* bar, float* outp, float* maxout) {
  int tid = blockIdx.x * BTH + threadIdx.x;   // grid 256x256 = 65536
  if (tid < BSZ * HDIM) {
    h0_32[tid] = 0.f; h1_32[tid] = 0.f;
    h0fA[tid] = (half_t)0.f; h0fB[tid] = (half_t)0.f;
    h1fA[tid] = (half_t)0.f; h1fB[tid] = (half_t)0.f;
  }
  if (tid < BSZ * VDIM) {
    int b = tid / VDIM, v = tid % VDIM;
    outp[((size_t)b * TSEQ) * VDIM + v] = (v == START_TOK) ? 1.f : 0.f;
  }
  if (tid < BSZ) {
    tok[tid] = START_TOK;
    maxout[(size_t)tid * TSEQ] = (float)START_TOK;
  }
  if (tid < 2) bar[tid] = 0;
}

// ---------------------------------------------------------------------------
extern "C" void kernel_launch(void* const* d_in, const int* in_sizes, int n_in,
                              void* d_out, int out_size, void* d_ws, size_t ws_size,
                              hipStream_t stream) {
  const float* input   = (const float*)d_in[0];
  const float* enc_emb = (const float*)d_in[1];
  const float* eWih0 = (const float*)d_in[2];
  const float* eWhh0 = (const float*)d_in[3];
  const float* ebih0 = (const float*)d_in[4];
  const float* ebhh0 = (const float*)d_in[5];
  const float* eWih1 = (const float*)d_in[6];
  const float* eWhh1 = (const float*)d_in[7];
  const float* ebih1 = (const float*)d_in[8];
  const float* ebhh1 = (const float*)d_in[9];
  const float* dec_emb = (const float*)d_in[10];
  const float* dWih0 = (const float*)d_in[11];
  const float* dWhh0 = (const float*)d_in[12];
  const float* dbih0 = (const float*)d_in[13];
  const float* dbhh0 = (const float*)d_in[14];
  const float* dWih1 = (const float*)d_in[15];
  const float* dWhh1 = (const float*)d_in[16];
  const float* dbih1 = (const float*)d_in[17];
  const float* dbhh1 = (const float*)d_in[18];
  const float* W1 = (const float*)d_in[19];
  const float* b1 = (const float*)d_in[20];
  const float* W2 = (const float*)d_in[21];
  const float* b2 = (const float*)d_in[22];

  float* outp   = (float*)d_out;                                // [B,T,V]
  float* maxout = outp + (size_t)BSZ * TSEQ * VDIM;             // [B,T]

  // ---- workspace carve-up (256B aligned) ----
  char* ws = (char*)d_ws;
  size_t off = 0;
  auto carve = [&](size_t bytes) {
    size_t o = off; off += (bytes + 255) & ~(size_t)255; return o;
  };
  int*    bar    = (int*)   (ws + carve(256));
  int*    tok    = (int*)   (ws + carve(BSZ * 4));
  float*  h0_32  = (float*) (ws + carve(BSZ * HDIM * 4));
  float*  h1_32  = (float*) (ws + carve(BSZ * HDIM * 4));
  half_t* h0fA   = (half_t*)(ws + carve(BSZ * HDIM * 2));
  half_t* h0fB   = (half_t*)(ws + carve(BSZ * HDIM * 2));
  half_t* h1fA   = (half_t*)(ws + carve(BSZ * HDIM * 2));
  half_t* h1fB   = (half_t*)(ws + carve(BSZ * HDIM * 2));
  half_t* hidf   = (half_t*)(ws + carve(BSZ * HDIM * 2));
  float*  logits = (float*) (ws + carve(BSZ * 400 * 4));
  half_t* pw_eWih0 = (half_t*)(ws + carve((size_t) 8 * 96 * 1024));
  half_t* pw_eWhh0 = (half_t*)(ws + carve((size_t)16 * 96 * 1024));
  half_t* pw_eWih1 = (half_t*)(ws + carve((size_t)16 * 96 * 1024));
  half_t* pw_eWhh1 = (half_t*)(ws + carve((size_t)16 * 96 * 1024));
  half_t* pw_dWih0 = (half_t*)(ws + carve((size_t) 8 * 96 * 1024));
  half_t* pw_dWhh0 = (half_t*)(ws + carve((size_t)16 * 96 * 1024));
  half_t* pw_dWih1 = (half_t*)(ws + carve((size_t)16 * 96 * 1024));
  half_t* pw_dWhh1 = (half_t*)(ws + carve((size_t)16 * 96 * 1024));
  half_t* pw_W1    = (half_t*)(ws + carve((size_t)16 * 32 * 1024));
  half_t* pw_W2    = (half_t*)(ws + carve((size_t)16 * 25 * 1024));
  half_t* xfrag    = (half_t*)(ws + carve((size_t)BSZ * TSEQ * EDIM * 2));
  (void)ws_size; (void)in_sizes; (void)n_in; (void)out_size;

  // 1) init state/barrier/t=0 outputs
  init_kernel<<<256, BTH, 0, stream>>>(h0_32, h1_32, h0fA, h0fB, h1fA, h1fB,
                                       tok, bar, outp, maxout);
  // 2) pack all weights into WMMA B fragments (f16, zero-padded)
  pack_b_kernel<<<768, BTH, 0, stream>>>(eWih0, pw_eWih0, G3H, EDIM, 96,  8);
  pack_b_kernel<<<768, BTH, 0, stream>>>(eWhh0, pw_eWhh0, G3H, HDIM, 96, 16);
  pack_b_kernel<<<768, BTH, 0, stream>>>(eWih1, pw_eWih1, G3H, HDIM, 96, 16);
  pack_b_kernel<<<768, BTH, 0, stream>>>(eWhh1, pw_eWhh1, G3H, HDIM, 96, 16);
  pack_b_kernel<<<768, BTH, 0, stream>>>(dWih0, pw_dWih0, G3H, EDIM, 96,  8);
  pack_b_kernel<<<768, BTH, 0, stream>>>(dWhh0, pw_dWhh0, G3H, HDIM, 96, 16);
  pack_b_kernel<<<768, BTH, 0, stream>>>(dWih1, pw_dWih1, G3H, HDIM, 96, 16);
  pack_b_kernel<<<768, BTH, 0, stream>>>(dWhh1, pw_dWhh1, G3H, HDIM, 96, 16);
  pack_b_kernel<<<768, BTH, 0, stream>>>(W1,    pw_W1,    HDIM, HDIM, 32, 16);
  pack_b_kernel<<<768, BTH, 0, stream>>>(W2,    pw_W2,    VDIM, HDIM, 25, 16);
  // 3) embedding GEMM -> encoder A-fragments for all T
  embed_kernel<<<BSZ * TSEQ, BTH, 0, stream>>>(input, enc_emb, xfrag);
  // 4) persistent sequential encoder + autoregressive decoder
  SeqArgs a;
  a.xfrag = xfrag;
  a.h0fA = h0fA; a.h0fB = h0fB; a.h1fA = h1fA; a.h1fB = h1fB; a.hidf = hidf;
  a.h0_32 = h0_32; a.h1_32 = h1_32; a.logits = logits;
  a.pw_eWih0 = pw_eWih0; a.pw_eWhh0 = pw_eWhh0;
  a.pw_eWih1 = pw_eWih1; a.pw_eWhh1 = pw_eWhh1;
  a.pw_dWih0 = pw_dWih0; a.pw_dWhh0 = pw_dWhh0;
  a.pw_dWih1 = pw_dWih1; a.pw_dWhh1 = pw_dWhh1;
  a.pw_W1 = pw_W1; a.pw_W2 = pw_W2;
  a.ebih0 = ebih0; a.ebhh0 = ebhh0; a.ebih1 = ebih1; a.ebhh1 = ebhh1;
  a.dbih0 = dbih0; a.dbhh0 = dbhh0; a.dbih1 = dbih1; a.dbhh1 = dbhh1;
  a.dec_emb = dec_emb; a.b1 = b1; a.b2 = b2;
  a.tok = tok; a.bar = bar; a.outp = outp; a.maxout = maxout;
  seq_kernel<<<NB, BTH, 0, stream>>>(a);
}
